// Model_55113020343030
// MI455X (gfx1250) — matrix operand order repaired and checked by reading, not compiled, not run on hardware
//
#include <hip/hip_runtime.h>
#include <math.h>

namespace {

constexpr int BATCH = 32, LENC = 512, LDEC = 592, DM = 512, DFF = 2048;
constexpr int NC = 7, NM = 4, PRED = 336, LABEL = 256;
constexpr int FFN_CHUNK = 4096;

typedef __attribute__((ext_vector_type(16))) __bf16 v16bf;
typedef __attribute__((ext_vector_type(2)))  __bf16 v2bf;
typedef __attribute__((ext_vector_type(8)))  float v8f;
typedef __attribute__((ext_vector_type(4)))  unsigned int u32x4;
typedef __attribute__((ext_vector_type(8)))  int i32x8;
typedef __attribute__((ext_vector_type(4)))  int i32x4;

union FragU { v16bf v; unsigned u[8]; };

__device__ __forceinline__ unsigned pack2bf(float a, float b) {
  v2bf t;                        // packed build -> v_cvt_pk_bf16_f32(a, b)
  t.x = (__bf16)a;
  t.y = (__bf16)b;
  return __builtin_bit_cast(unsigned, t);
}

// ---------------------------------------------------------------------------
// WMMA bf16 GEMM: C[bz] = act( A[bz] (MxK, f32) * B[bz]^T + bias )
//   btype 0: B is f32, rows are output-cols (N x K row-major)  [Q*K^T path]
//            staged by vector loads + convert.
//   btype 1: B is bf16, pre-transposed N x K row-major          [weights]
//            staged by the Tensor Data Mover: 32x128 2-B tile with
//            pad_enable (4 dwords per 16 dwords) -> LDS row stride 80B.
// Rows of B >= nvalid are treated as zero (zero-padded K/V for cross attn).
// Block tile 128x128x32, 8 waves (4x2), each wave 2x4 WMMA 16x16x32 tiles.
// Double-buffered LDS + register/TDM prefetch; one barrier per k-step.
// ---------------------------------------------------------------------------
__global__ __launch_bounds__(256) void gemm_wmma_bf16(
    const float* __restrict__ A, const void* __restrict__ Bv,
    const float* __restrict__ bias, float* __restrict__ C,
    int M, int N, int K, int lda, int ldb, int ldc,
    long long strA, long long strB, long long strC,
    int btype, int nvalid, int act)
{
  constexpr int LDK = 40;                       // ushorts per LDS row (80B)
  __shared__ __align__(16) unsigned short shA[2][128 * LDK];
  __shared__ __align__(16) unsigned short shB[2][128 * LDK];
  const int bz = blockIdx.z;
  A += strA * bz;
  const float*          Bf = (const float*)Bv + strB * bz;
  const unsigned short* Bh = (const unsigned short*)Bv + strB * bz;
  C += strC * bz;
  const int tid  = threadIdx.x;
  const int lane = tid & 31, wave = tid >> 5;
  const int wm = wave >> 1, wn = wave & 1;
  const int m0 = blockIdx.y * 128, n0 = blockIdx.x * 128;
  const int arow = tid >> 3, acol = (tid & 7) * 4;   // f32 staging mapping

  const v8f vzero = {0.f,0.f,0.f,0.f,0.f,0.f,0.f,0.f};
  v8f acc[2][4];
#pragma unroll
  for (int i = 0; i < 2; i++)
#pragma unroll
    for (int j = 0; j < 4; j++) acc[i][j] = vzero;

  float4 ar[4];          // A prefetch regs
  float4 br[4];          // B prefetch regs (f32 path)

  auto loadA = [&](int k0) {
#pragma unroll
    for (int p = 0; p < 4; p++) {
      int gm = m0 + arow + p * 32;
      int gmc = gm < M ? gm : M - 1;               // clamp: branchless OOB
      float4 v = *(const float4*)(A + (long long)gmc * lda + k0 + acol);
      if (gm >= M) v = make_float4(0.f, 0.f, 0.f, 0.f);
      ar[p] = v;
    }
  };
  auto loadBf32 = [&](int k0) {
#pragma unroll
    for (int p = 0; p < 4; p++) {
      int gn = n0 + arow + p * 32;
      int gnc = gn < nvalid ? gn : nvalid - 1;
      float4 v = *(const float4*)(Bf + (long long)gnc * ldb + k0 + acol);
      if (gn >= nvalid) v = make_float4(0.f, 0.f, 0.f, 0.f);
      br[p] = v;
    }
  };
  auto stageA = [&](unsigned short* sh) {
#pragma unroll
    for (int p = 0; p < 4; p++)
      *(uint2*)&sh[(arow + p * 32) * LDK + acol] =
          make_uint2(pack2bf(ar[p].x, ar[p].y), pack2bf(ar[p].z, ar[p].w));
  };
  auto stageBf32 = [&](unsigned short* sh) {
#pragma unroll
    for (int p = 0; p < 4; p++)
      *(uint2*)&sh[(arow + p * 32) * LDK + acol] =
          make_uint2(pack2bf(br[p].x, br[p].y), pack2bf(br[p].z, br[p].w));
  };

  // TDM: async 2-D tile (32 k-halves x 128 n-rows) of bf16 B into LDS.
  // pad_interval=3 (16 dwords), pad_amount=3 (4 dwords) => 80B LDS rows.
  auto tdmIssueB = [&](int k0, unsigned short* dstLds) {
    unsigned lds_off = (unsigned)(unsigned long long)(void*)dstLds;
    unsigned long long ga =
        (unsigned long long)(Bh + (long long)n0 * ldb + k0);
    int td0 = K - k0;                      // k extent from tile start
    int td1 = nvalid - n0;                 // valid rows (OOB reads -> 0)
    u32x4 g0 = { 1u, lds_off, (unsigned)ga,
                 (unsigned)((ga >> 32) & 0x1FFFFFFull) | (2u << 30) };
    i32x8 g1 = {
      (int)((1u << 16) | (1u << 20) | (3u << 22) | (3u << 25)),
      (int)((unsigned)(td0 & 0xffff) << 16),
      (int)(((unsigned)(td0 >> 16) & 0xffffu) |
            ((unsigned)(td1 & 0xffff) << 16)),
      (int)(((unsigned)(td1 >> 16) & 0xffffu) | (32u << 16)),  // tile_dim0=32
      128,                                                      // tile_dim1
      ldb,                                                      // dim0 stride
      0, 0 };
    i32x4 g2 = { 0, 0, 0, 0 };
    i32x4 g3 = { 0, 0, 0, 0 };
    i32x8 g4 = { 0, 0, 0, 0, 0, 0, 0, 0 };
    __builtin_amdgcn_tensor_load_to_lds(g0, g1, g2, g3, g4, 0);
  };

  // prologue
  loadA(0);
  if (btype == 0) loadBf32(0);
  else if (wave == 0) tdmIssueB(0, shB[0]);

  int pp = 0;
  const int half = lane >> 4, l15 = lane & 15;
  for (int k0 = 0; k0 < K; k0 += 32) {
    stageA(shA[pp]);
    if (btype == 0) stageBf32(shB[pp]);
    else if (wave == 0) __builtin_amdgcn_s_wait_tensorcnt(0);
    __syncthreads();
    if (k0 + 32 < K) {                       // prefetch next tile
      loadA(k0 + 32);
      if (btype == 0) loadBf32(k0 + 32);
      else if (wave == 0) tdmIssueB(k0 + 32, shB[pp ^ 1]);
    }

    FragU afr[2], bfr[4];
    // A fragment: ISA 7.12.2 16-bit A 16x32: k = (v>>2)*16 + half*8 + (v&3)*2
#pragma unroll
    for (int tm = 0; tm < 2; tm++) {
      const int row = wm * 32 + tm * 16 + l15;
#pragma unroll
      for (int v = 0; v < 8; v++) {
        const int kk = (v >> 2) * 16 + half * 8 + (v & 3) * 2;
        afr[tm].u[v] = *(const unsigned*)&shA[pp][row * LDK + kk];
      }
    }
    // B fragment: same k order as A: k = (v>>2)*16 + half*8 + (v&3)*2
#pragma unroll
    for (int tn = 0; tn < 4; tn++) {
      const int col = wn * 64 + tn * 16 + l15;
#pragma unroll
      for (int v = 0; v < 8; v++) {
        const int kk = (v >> 2) * 16 + half * 8 + (v & 3) * 2;
        bfr[tn].u[v] = *(const unsigned*)&shB[pp][col * LDK + kk];
      }
    }
#pragma unroll
    for (int tm = 0; tm < 2; tm++)
#pragma unroll
      for (int tn = 0; tn < 4; tn++)
        acc[tm][tn] = __builtin_amdgcn_wmma_f32_16x16x32_bf16(
            false, afr[tm].v, false, bfr[tn].v, (short)0, acc[tm][tn],
            false, false);
    pp ^= 1;
  }

#pragma unroll
  for (int tm = 0; tm < 2; tm++)
#pragma unroll
    for (int tn = 0; tn < 4; tn++)
#pragma unroll
      for (int r = 0; r < 8; r++) {
        int row = m0 + wm * 32 + tm * 16 + r + half * 8;
        int col = n0 + wn * 64 + tn * 16 + l15;
        if (row < M && col < N) {
          float v = acc[tm][tn][r];
          if (bias) v += bias[col];
          if (act == 1) v = 0.5f * v * (1.0f + erff(v * 0.70710678118f));
          C[(long long)row * ldc + col] = v;
        }
      }
}

// weight f32 (KxN) -> bf16 transposed (NxK)
__global__ void k_w2bft(const float* __restrict__ src, unsigned short* __restrict__ dst,
                        int K, int N) {
  int i = blockIdx.x * blockDim.x + threadIdx.x;
  int tot = N * (K >> 2);
  if (i >= tot) return;
  int kq = i % (K >> 2), n = i / (K >> 2);
  int k = kq * 4;
  float f0 = src[(long long)(k + 0) * N + n];
  float f1 = src[(long long)(k + 1) * N + n];
  float f2 = src[(long long)(k + 2) * N + n];
  float f3 = src[(long long)(k + 3) * N + n];
  *(uint2*)&dst[(long long)n * K + k] =
      make_uint2(pack2bf(f0, f1), pack2bf(f2, f3));
}

// ---------------------------------------------------------------------------
// Elementwise / small kernels
// ---------------------------------------------------------------------------
__global__ void k_mean_time(const float* __restrict__ x, float* __restrict__ out,
                            int B, int L, int C) {
  int i = blockIdx.x * blockDim.x + threadIdx.x;
  if (i >= B * C) return;
  int b = i / C, c = i % C;
  float s = 0.f;
  for (int t = 0; t < L; t++) s += x[((long long)b * L + t) * C + c];
  out[i] = s / (float)L;
}

__global__ void k_decomp(const float* __restrict__ x, float* __restrict__ seas,
                         float* __restrict__ trend, int B, int L, int C) {
  long long i = (long long)blockIdx.x * blockDim.x + threadIdx.x;
  long long tot = (long long)B * L * C;
  if (i >= tot) return;
  int c = (int)(i % C);
  int t = (int)((i / C) % L);
  int b = (int)(i / ((long long)C * L));
  float s = 0.f;
  for (int j = -12; j <= 12; j++) {
    int tt = t + j;
    tt = tt < 0 ? 0 : (tt >= L ? L - 1 : tt);
    s += x[((long long)b * L + tt) * C + c];
  }
  float m = s * (1.0f / 25.0f);
  float v = x[i];
  seas[i] = v - m;
  trend[i] = m;
}

__global__ void k_add(const float* __restrict__ a, const float* __restrict__ b,
                      float* __restrict__ c, long long n) {
  long long i = (long long)blockIdx.x * blockDim.x + threadIdx.x;
  if (i < n) c[i] = a[i] + b[i];
}

__global__ void k_embed(const float* __restrict__ x, const float* __restrict__ mark,
                        const float* __restrict__ Wc, const float* __restrict__ Wm,
                        float* __restrict__ out, int B, int L, int Cin, int Mm) {
  long long i = (long long)blockIdx.x * blockDim.x + threadIdx.x;
  long long tot = (long long)B * L * DM;
  if (i >= tot) return;
  int j = (int)(i % DM);
  int t = (int)((i / DM) % L);
  int b = (int)(i / ((long long)DM * L));
  float s = 0.f;
  for (int w = 0; w < 3; w++) {
    int tt = (t - 1 + w + L) % L;
    const float* xr = &x[((long long)b * L + tt) * Cin];
    for (int c = 0; c < Cin; c++) s += xr[c] * Wc[(w * Cin + c) * DM + j];
  }
  const float* mr = &mark[((long long)b * L + t) * Mm];
  for (int m = 0; m < Mm; m++) s += mr[m] * Wm[m * DM + j];
  out[i] = s;
}

__global__ void k_diag_mean(const float* __restrict__ G, float* __restrict__ mc,
                            int B, int L) {
  int i = blockIdx.x * blockDim.x + threadIdx.x;
  if (i >= B * L) return;
  int b = i / L, tau = i % L;
  const float* Gb = G + (long long)b * L * L;
  float s = 0.f;
  for (int j = 0; j < L; j++) {
    int r = j + tau; if (r >= L) r -= L;
    s += Gb[(long long)r * L + j];
  }
  mc[i] = s * (1.0f / (float)DM);
}

__global__ void k_topk(const float* __restrict__ mc, float* __restrict__ w,
                       int* __restrict__ delay, int L, int k) {
  int b = blockIdx.x, lane = threadIdx.x;
  const float* row = mc + (long long)b * L;
  int sel[20]; float vals[20];
  for (int i = 0; i < k; i++) {
    float best = -3.4e38f; int bi = -1;
    for (int t = lane; t < L; t += 32) {
      bool used = false;
      for (int j = 0; j < i; j++) if (sel[j] == t) used = true;
      float v = row[t];
      if (!used && v > best) { best = v; bi = t; }
    }
    for (int off = 16; off > 0; off >>= 1) {
      float ov = __shfl_down(best, off);
      int   oi = __shfl_down(bi, off);
      if (oi >= 0 && ov > best) { best = ov; bi = oi; }
    }
    best = __shfl(best, 0); bi = __shfl(bi, 0);
    sel[i] = bi; vals[i] = best;
  }
  if (lane == 0) {
    float mx = vals[0];
    for (int i = 1; i < k; i++) mx = fmaxf(mx, vals[i]);
    float se = 0.f; float e[20];
    for (int i = 0; i < k; i++) { e[i] = expf(vals[i] - mx); se += e[i]; }
    float inv = 1.0f / se;
    for (int i = 0; i < k; i++) { w[b * 32 + i] = e[i] * inv; delay[b * 32 + i] = sel[i]; }
  }
}

__global__ void k_agg(const float* __restrict__ v, const float* __restrict__ w,
                      const int* __restrict__ delay, float* __restrict__ out,
                      int B, int L, int Sv, int k) {
  long long i = (long long)blockIdx.x * blockDim.x + threadIdx.x;
  long long tot = (long long)B * L * DM;
  if (i >= tot) return;
  int d = (int)(i % DM);
  int t = (int)((i / DM) % L);
  int b = (int)(i / ((long long)DM * L));
  float s = 0.f;
  for (int j = 0; j < k; j++) {
    int idx = t + delay[b * 32 + j]; if (idx >= L) idx -= L;
    if (idx < Sv) s += w[b * 32 + j] * v[((long long)b * Sv + idx) * DM + d];
  }
  out[i] = s;
}

__global__ __launch_bounds__(256) void k_ln(const float* __restrict__ x,
                                            const float* __restrict__ g,
                                            const float* __restrict__ be,
                                            float* __restrict__ out, int rows) {
  int r = blockIdx.x * 8 + (threadIdx.x >> 5);
  int lane = threadIdx.x & 31;
  if (r >= rows) return;
  const float* xr = x + (long long)r * DM;
  float s = 0.f, s2 = 0.f;
  for (int j = lane; j < DM; j += 32) { float v = xr[j]; s += v; s2 += v * v; }
  for (int off = 16; off > 0; off >>= 1) { s += __shfl_xor(s, off); s2 += __shfl_xor(s2, off); }
  float mu = s / (float)DM;
  float var = s2 / (float)DM - mu * mu;
  float inv = rsqrtf(var + 1e-5f);
  float* orow = out + (long long)r * DM;
  for (int j = lane; j < DM; j += 32) orow[j] = (xr[j] - mu) * inv * g[j] + be[j];
}

__global__ void k_sub_colmean(const float* __restrict__ x, float* __restrict__ out,
                              int B, int L) {
  int i = blockIdx.x * blockDim.x + threadIdx.x;
  if (i >= B * DM) return;
  int b = i / DM, d = i % DM;
  float s = 0.f;
  for (int t = 0; t < L; t++) s += x[((long long)b * L + t) * DM + d];
  float m = s / (float)L;
  for (int t = 0; t < L; t++) {
    long long o = ((long long)b * L + t) * DM + d;
    out[o] = x[o] - m;
  }
}

__global__ void k_builddec(const float* __restrict__ se, const float* __restrict__ te,
                           const float* __restrict__ meanc,
                           const float* __restrict__ xme, const float* __restrict__ xmd,
                           float* __restrict__ sinit, float* __restrict__ tinit,
                           float* __restrict__ mdec) {
  int i = blockIdx.x * blockDim.x + threadIdx.x;
  if (i >= BATCH * LDEC) return;
  int b = i / LDEC, t = i % LDEC;
  for (int c = 0; c < NC; c++) {
    long long o = ((long long)b * LDEC + t) * NC + c;
    if (t < LABEL) {
      long long s = ((long long)b * LENC + (LENC - LABEL + t)) * NC + c;
      sinit[o] = se[s];
      tinit[o] = te[s];
    } else {
      sinit[o] = 0.f;
      tinit[o] = meanc[b * NC + c];
    }
  }
  for (int m = 0; m < NM; m++) {
    long long o = ((long long)b * LDEC + t) * NM + m;
    mdec[o] = (t < LABEL)
        ? xme[((long long)b * LENC + (LENC - LABEL + t)) * NM + m]
        : xmd[((long long)b * PRED + (t - LABEL)) * NM + m];
  }
}

__global__ void k_trend_add(const float* __restrict__ tinit, const float* __restrict__ tsum,
                            const float* __restrict__ Wt, float* __restrict__ out) {
  int i = blockIdx.x * blockDim.x + threadIdx.x;
  if (i >= BATCH * LDEC * NC) return;
  int c = i % NC;
  int t = (i / NC) % LDEC;
  int b = i / (NC * LDEC);
  float s = tinit[i];
  for (int w = 0; w < 3; w++) {
    int tt = (t - 1 + w + LDEC) % LDEC;
    const float* row = tsum + ((long long)b * LDEC + tt) * DM;
    for (int d = 0; d < DM; d++) s += row[d] * Wt[(w * DM + d) * NC + c];
  }
  out[i] = s;
}

__global__ void k_final(const float* __restrict__ trend, const float* __restrict__ x,
                        const float* __restrict__ Wp, const float* __restrict__ bp,
                        float* __restrict__ out) {
  int i = blockIdx.x * blockDim.x + threadIdx.x;
  if (i >= BATCH * PRED * NC) return;
  int c = i % NC;
  int t = (i / NC) % PRED;
  int b = i / (NC * PRED);
  int tt = LABEL + t;
  float s = bp[c] + trend[((long long)b * LDEC + tt) * NC + c];
  const float* xr = x + ((long long)b * LDEC + tt) * DM;
  for (int d = 0; d < DM; d++) s += xr[d] * Wp[d * NC + c];
  out[i] = s;
}

} // namespace

// ---------------------------------------------------------------------------
extern "C" void kernel_launch(void* const* d_in, const int* in_sizes, int n_in,
                              void* d_out, int out_size, void* d_ws, size_t ws_size,
                              hipStream_t stream) {
  (void)in_sizes; (void)n_in; (void)out_size; (void)ws_size;
  const float* x_enc      = (const float*)d_in[0];
  const float* x_mark_enc = (const float*)d_in[1];
  const float* x_mark_dec = (const float*)d_in[3];
  const float* W_enc_emb  = (const float*)d_in[4];
  const float* W_mark_enc = (const float*)d_in[5];
  const float* W_dec_emb  = (const float*)d_in[6];
  const float* W_mark_dec = (const float*)d_in[7];
  const float* eW_attn    = (const float*)d_in[8];
  const float* eb_attn    = (const float*)d_in[9];
  const float* eW1        = (const float*)d_in[10];
  const float* eW2        = (const float*)d_in[11];
  const float* eg         = (const float*)d_in[12];
  const float* ebeta      = (const float*)d_in[13];
  const float* sW         = (const float*)d_in[14];
  const float* sb         = (const float*)d_in[15];
  const float* cW         = (const float*)d_in[16];
  const float* cb         = (const float*)d_in[17];
  const float* dW1        = (const float*)d_in[18];
  const float* dW2        = (const float*)d_in[19];
  const float* W_trend    = (const float*)d_in[20];
  const float* dg         = (const float*)d_in[21];
  const float* dbeta      = (const float*)d_in[22];
  const float* Wproj      = (const float*)d_in[23];
  const float* bproj      = (const float*)d_in[24];
  float* out = (float*)d_out;

  // ---- workspace carving
  char* p = (char*)d_ws;
  auto alloc = [&](size_t bytes) {
    void* r = (void*)p;
    p += (bytes + 255) & ~(size_t)255;
    return r;
  };
  const size_t SZ_BIG = (size_t)BATCH * LDEC * DM * 4;
  float* X    = (float*)alloc(SZ_BIG);
  float* S    = (float*)alloc(SZ_BIG);
  float* T    = (float*)alloc(SZ_BIG);
  float* TSUM = (float*)alloc(SZ_BIG);
  float* Qb   = (float*)alloc(SZ_BIG);
  float* Kb   = (float*)alloc(SZ_BIG);
  float* Vb   = (float*)alloc(SZ_BIG);
  float* Gb   = (float*)alloc((size_t)BATCH * LDEC * LDEC * 4);
  float* ENC  = (float*)alloc((size_t)BATCH * LENC * DM * 4);
  float* FFNT = (float*)alloc((size_t)FFN_CHUNK * DFF * 4);
  float* SE   = (float*)alloc((size_t)BATCH * LENC * NC * 4);
  float* TE   = (float*)alloc((size_t)BATCH * LENC * NC * 4);
  float* SINIT  = (float*)alloc((size_t)BATCH * LDEC * NC * 4);
  float* TINIT  = (float*)alloc((size_t)BATCH * LDEC * NC * 4);
  float* TRFULL = (float*)alloc((size_t)BATCH * LDEC * NC * 4);
  float* MDEC = (float*)alloc((size_t)BATCH * LDEC * NM * 4);
  float* MEANC = (float*)alloc((size_t)BATCH * NC * 4);
  float* MC   = (float*)alloc((size_t)BATCH * LDEC * 4);
  float* WK   = (float*)alloc((size_t)BATCH * 32 * 4);
  int*   DEL  = (int*)  alloc((size_t)BATCH * 32 * 4);
  // bf16 pre-transposed weights
  unsigned short* WB_eatt = (unsigned short*)alloc((size_t)2 * 4 * DM * DM * 2);
  unsigned short* WB_sW   = (unsigned short*)alloc((size_t)4 * DM * DM * 2);
  unsigned short* WB_cW   = (unsigned short*)alloc((size_t)4 * DM * DM * 2);
  unsigned short* WB_e1   = (unsigned short*)alloc((size_t)2 * DM * DFF * 2);
  unsigned short* WB_e2   = (unsigned short*)alloc((size_t)2 * DFF * DM * 2);
  unsigned short* WB_d1   = (unsigned short*)alloc((size_t)DM * DFF * 2);
  unsigned short* WB_d2   = (unsigned short*)alloc((size_t)DFF * DM * 2);

  auto blks = [](long long n) { return dim3((unsigned)((n + 255) / 256)); };
  dim3 th(256);

  auto wcvt = [&](const float* s, unsigned short* d, int K, int N) {
    k_w2bft<<<blks((long long)N * (K / 4)), th, 0, stream>>>(s, d, K, N);
  };
  for (int l = 0; l < 2; l++)
    for (int j = 0; j < 4; j++)
      wcvt(eW_attn + (size_t)(l * 4 + j) * DM * DM, WB_eatt + (size_t)(l * 4 + j) * DM * DM, DM, DM);
  for (int j = 0; j < 4; j++) {
    wcvt(sW + (size_t)j * DM * DM, WB_sW + (size_t)j * DM * DM, DM, DM);
    wcvt(cW + (size_t)j * DM * DM, WB_cW + (size_t)j * DM * DM, DM, DM);
  }
  for (int l = 0; l < 2; l++) {
    wcvt(eW1 + (size_t)l * DM * DFF, WB_e1 + (size_t)l * DM * DFF, DM, DFF);
    wcvt(eW2 + (size_t)l * DFF * DM, WB_e2 + (size_t)l * DFF * DM, DFF, DM);
  }
  wcvt(dW1, WB_d1, DM, DFF);
  wcvt(dW2, WB_d2, DFF, DM);

  auto gemm = [&](const float* A, const void* B, const float* bias, float* C,
                  int M, int N, int K, int lda, int ldb, int ldc,
                  long long sa, long long sb_, long long sc, int nb,
                  int btype, int nvalid, int act) {
    dim3 g((N + 127) / 128, (M + 127) / 128, nb);
    gemm_wmma_bf16<<<g, th, 0, stream>>>(A, B, bias, C, M, N, K, lda, ldb, ldc,
                                         sa, sb_, sc, btype, nvalid, act);
  };

  // autocorrelation attention; output -> outp (Kb is dead by then)
  auto attn = [&](const float* qin, int Lq, const float* kin, int Lkv,
                  const unsigned short* Wb, const float* b4, float* outp) {
    int Mq = BATCH * Lq, Mk = BATCH * Lkv;
    gemm(qin, Wb + 0ull * DM * DM, b4 + 0 * DM, Qb, Mq, DM, DM, DM, DM, DM, 0, 0, 0, 1, 1, DM, 0);
    gemm(kin, Wb + 1ull * DM * DM, b4 + 1 * DM, Kb, Mk, DM, DM, DM, DM, DM, 0, 0, 0, 1, 1, DM, 0);
    gemm(kin, Wb + 2ull * DM * DM, b4 + 2 * DM, Vb, Mk, DM, DM, DM, DM, DM, 0, 0, 0, 1, 1, DM, 0);
    // G[b] = Q[b] (Lq x DM) * K[b]^T   (rows >= Lkv are zero)
    gemm(Qb, Kb, nullptr, Gb, Lq, Lq, DM, DM, DM, Lq,
         (long long)Lq * DM, (long long)Lkv * DM, (long long)Lq * Lq, BATCH, 0, Lkv, 0);
    int tk = (Lq == LENC) ? 18 : 19;           // int(3*ln(L))
    k_diag_mean<<<blks((long long)BATCH * Lq), th, 0, stream>>>(Gb, MC, BATCH, Lq);
    k_topk<<<dim3(BATCH), dim3(32), 0, stream>>>(MC, WK, DEL, Lq, tk);
    k_agg<<<blks((long long)BATCH * Lq * DM), th, 0, stream>>>(Vb, WK, DEL, Qb, BATCH, Lq, Lkv, tk);
    gemm(Qb, Wb + 3ull * DM * DM, b4 + 3 * DM, outp, Mq, DM, DM, DM, DM, DM, 0, 0, 0, 1, 1, DM, 0);
  };

  auto ffn = [&](const unsigned short* W1b, const unsigned short* W2b, int rows) {
    for (int off = 0; off < rows; off += FFN_CHUNK) {
      int mm = rows - off; if (mm > FFN_CHUNK) mm = FFN_CHUNK;
      gemm(X + (long long)off * DM, W1b, nullptr, FFNT, mm, DFF, DM, DM, DM, DFF,
           0, 0, 0, 1, 1, DFF, 1 /*gelu*/);
      gemm(FFNT, W2b, nullptr, Kb + (long long)off * DM, mm, DM, DFF, DFF, DFF, DM,
           0, 0, 0, 1, 1, DM, 0);
    }
  };

  // ================= encoder =================
  k_embed<<<blks((long long)BATCH * LENC * DM), th, 0, stream>>>(
      x_enc, x_mark_enc, W_enc_emb, W_mark_enc, X, BATCH, LENC, NC, NM);

  const long long nEnc = (long long)BATCH * LENC * DM;
  for (int l = 0; l < 2; l++) {
    attn(X, LENC, X, LENC, WB_eatt + (size_t)l * 4 * DM * DM,
         eb_attn + (size_t)l * 4 * DM, Kb);
    k_add<<<blks(nEnc), th, 0, stream>>>(X, Kb, S, nEnc);
    k_decomp<<<blks(nEnc), th, 0, stream>>>(S, X, T, BATCH, LENC, DM);
    ffn(WB_e1 + (size_t)l * DM * DFF, WB_e2 + (size_t)l * DFF * DM, BATCH * LENC);
    k_add<<<blks(nEnc), th, 0, stream>>>(X, Kb, S, nEnc);
    k_decomp<<<blks(nEnc), th, 0, stream>>>(S, X, T, BATCH, LENC, DM);
  }
  k_ln<<<dim3((BATCH * LENC + 7) / 8), th, 0, stream>>>(X, eg, ebeta, S, BATCH * LENC);
  k_sub_colmean<<<blks(BATCH * DM), th, 0, stream>>>(S, ENC, BATCH, LENC);

  // ================= decoder inputs =================
  k_mean_time<<<blks(BATCH * NC), th, 0, stream>>>(x_enc, MEANC, BATCH, LENC, NC);
  k_decomp<<<blks((long long)BATCH * LENC * NC), th, 0, stream>>>(x_enc, SE, TE, BATCH, LENC, NC);
  k_builddec<<<blks(BATCH * LDEC), th, 0, stream>>>(SE, TE, MEANC, x_mark_enc, x_mark_dec,
                                                    SINIT, TINIT, MDEC);
  k_embed<<<blks((long long)BATCH * LDEC * DM), th, 0, stream>>>(
      SINIT, MDEC, W_dec_emb, W_mark_dec, X, BATCH, LDEC, NC, NM);

  // ================= decoder =================
  const long long nDec = (long long)BATCH * LDEC * DM;
  attn(X, LDEC, X, LDEC, WB_sW, sb, Kb);                    // self attn
  k_add<<<blks(nDec), th, 0, stream>>>(X, Kb, S, nDec);
  k_decomp<<<blks(nDec), th, 0, stream>>>(S, X, TSUM, BATCH, LDEC, DM);   // t1

  attn(X, LDEC, ENC, LENC, WB_cW, cb, Kb);                  // cross attn (S=512 < L=592)
  k_add<<<blks(nDec), th, 0, stream>>>(X, Kb, S, nDec);
  k_decomp<<<blks(nDec), th, 0, stream>>>(S, X, T, BATCH, LDEC, DM);      // t2
  k_add<<<blks(nDec), th, 0, stream>>>(TSUM, T, TSUM, nDec);

  ffn(WB_d1, WB_d2, BATCH * LDEC);
  k_add<<<blks(nDec), th, 0, stream>>>(X, Kb, S, nDec);
  k_decomp<<<blks(nDec), th, 0, stream>>>(S, X, T, BATCH, LDEC, DM);      // t3
  k_add<<<blks(nDec), th, 0, stream>>>(TSUM, T, TSUM, nDec);

  k_trend_add<<<blks(BATCH * LDEC * NC), th, 0, stream>>>(TINIT, TSUM, W_trend, TRFULL);
  k_ln<<<dim3((BATCH * LDEC + 7) / 8), th, 0, stream>>>(X, dg, dbeta, S, BATCH * LDEC);
  k_sub_colmean<<<blks(BATCH * DM), th, 0, stream>>>(S, Kb, BATCH, LDEC);
  k_final<<<blks(BATCH * PRED * NC), th, 0, stream>>>(TRFULL, Kb, Wproj, bproj, out);
}
